// GARNet_1606317769306
// MI455X (gfx1250) — matrix operand order repaired and checked by reading, not compiled, not run on hardware
//
#include <hip/hip_runtime.h>

// ---------------------------------------------------------------------------
// Sizes fixed by the reference
// ---------------------------------------------------------------------------
#define NN   4096
#define DD   512
#define SS   512
#define EE   1024              // emb width
#define HOUT 512               // head layer-1 width

typedef __attribute__((ext_vector_type(16))) __bf16     v16bf;
typedef __attribute__((ext_vector_type(8)))  float      v8f;
typedef __attribute__((ext_vector_type(4)))  unsigned   u32x4;
typedef __attribute__((ext_vector_type(4)))  float      f32x4;

union V16U { v16bf v; u32x4 q[2]; };

__device__ __forceinline__ unsigned short f2bf(float f) {
    unsigned u = __float_as_uint(f);
    unsigned r = u + 0x7FFFu + ((u >> 16) & 1u);   // round-to-nearest-even
    return (unsigned short)(r >> 16);
}
__device__ __forceinline__ float bf2f(unsigned short h) {
    return __uint_as_float(((unsigned)h) << 16);
}
// order-preserving float <-> uint map for deterministic atomic max
__device__ __forceinline__ unsigned flipf(float f) {
    unsigned u = __float_as_uint(f);
    return (u & 0x80000000u) ? ~u : (u | 0x80000000u);
}
__device__ __forceinline__ float unflipf(unsigned u) {
    unsigned v = (u & 0x80000000u) ? (u & 0x7FFFFFFFu) : ~u;
    return __uint_as_float(v);
}

__device__ __forceinline__ v8f wmma_bf16(const V16U& a, const V16U& b, v8f c) {
    return __builtin_amdgcn_wmma_f32_16x16x32_bf16(
               false, a.v, false, b.v, (short)0, c, false, false);
}

// ---------------------------------------------------------------------------
// 0) zero the pooled-max (flipped) buffer
// ---------------------------------------------------------------------------
__global__ __launch_bounds__(256) void k_zero_u32(unsigned* p, int n) {
    int i = blockIdx.x * 256 + threadIdx.x;
    if (i < n) p[i] = 0u;     // flip-order minimum
}

// 1) segment max via order-independent uint atomicMax
__global__ __launch_bounds__(256) void k_segmax(const float* __restrict__ x,
                                                const int* __restrict__ seg,
                                                unsigned* __restrict__ pool) {
    int n = blockIdx.x, t = threadIdx.x;
    int s = seg[n];
    for (int d = t; d < DD; d += 256)
        atomicMax(&pool[(size_t)s * DD + d], flipf(x[(size_t)n * DD + d]));
}

// 2) build emb (bf16) = [x | pooled[seg]] and sqnorm of bf16-rounded x
__global__ __launch_bounds__(256) void k_build_emb(const float* __restrict__ x,
                                                   const unsigned* __restrict__ pool,
                                                   const int* __restrict__ seg,
                                                   unsigned short* __restrict__ emb,
                                                   float* __restrict__ sq) {
    __shared__ float red[256];
    int n = blockIdx.x, t = threadIdx.x;
    int s = seg[n];
    float local = 0.f;
    for (int d = t; d < DD; d += 256) {
        float v = x[(size_t)n * DD + d];
        unsigned short bv = f2bf(v);
        emb[(size_t)n * EE + d] = bv;
        float vb = bf2f(bv);
        local += vb * vb;
        emb[(size_t)n * EE + DD + d] = f2bf(unflipf(pool[(size_t)s * DD + d]));
    }
    red[t] = local;
    __syncthreads();
    for (int st = 128; st > 0; st >>= 1) {
        if (t < st) red[t] += red[t + st];
        __syncthreads();
    }
    if (t == 0) sq[n] = red[0];
}

// 3) convert head_w1 f32 -> bf16
__global__ __launch_bounds__(256) void k_f2bf(const float* __restrict__ w,
                                              unsigned short* __restrict__ wb, int n) {
    int i = blockIdx.x * 256 + threadIdx.x;
    if (i < n) wb[i] = f2bf(w[i]);
}

// ---------------------------------------------------------------------------
// 4) pairwise kernel: dist = cdist(x) via bf16 WMMA, fused GIoU + sigmoid.
//    Block tile 128(m) x 64(n): 8 waves in a 4x2 grid, each wave a 32x32
//    tile (2x2 WMMA fragments). The 64-row B strip (shared by all waves) is
//    staged per 128-K chunk into LDS via async global->LDS copies.
// ---------------------------------------------------------------------------
#define KC 128

__global__ __launch_bounds__(256) void k_pair_sim(const unsigned short* __restrict__ xbf, // emb[:, :512], row stride EE
                                                  const float* __restrict__ sq,
                                                  const float* __restrict__ boxes,
                                                  const float* __restrict__ dw,
                                                  const float* __restrict__ db,
                                                  float* __restrict__ out) {
    __shared__ __attribute__((aligned(16))) unsigned short smemB[64 * KC]; // 16 KB

    int t    = threadIdx.x;
    int wave = t >> 5;
    int lane = t & 31;
    int l16  = lane & 15;
    int hi   = lane >> 4;

    int bm = blockIdx.x >> 6;            // 32 m-blocks of 128
    int bn = blockIdx.x & 63;            // 64 n-blocks of 64
    int mBase = bm * 128, nBase = bn * 64;
    int wm = wave >> 1, wn = wave & 1;   // 4x2 wave grid
    int m0w = mBase + wm * 32;
    int n0w = nBase + wn * 32;

    // A fragment row pointers (ISA 16-bit A 16x32 layout: row = l16,
    // K chunks at (hi?8:0) and +16)
    const unsigned short* arow0 = xbf + (size_t)(m0w + l16) * EE + (hi ? 8 : 0);
    const unsigned short* arow1 = xbf + (size_t)(m0w + 16 + l16) * EE + (hi ? 8 : 0);

    v8f acc[2][2] = {};

    for (int kc = 0; kc < DD; kc += KC) {
        __syncthreads();   // previous chunk fully consumed before overwrite
        // --- stage B strip chunk: 64 rows x 128 K (bf16) via async copies ---
        #pragma unroll
        for (int i = 0; i < 4; ++i) {
            int slot = t + 256 * i;            // 1024 x 16B slots
            int row  = slot >> 4;
            int kp   = (slot & 15) << 3;       // 8 bf16 per slot
            const unsigned short* gp = xbf + (size_t)(nBase + row) * EE + kc + kp;
            unsigned lds_off = (unsigned)(unsigned long long)(&smemB[row * KC + kp]);
            unsigned long long ga = (unsigned long long)gp;
            asm volatile("global_load_async_to_lds_b128 %0, %1, off"
                         :: "v"(lds_off), "v"(ga) : "memory");
        }
        asm volatile("s_wait_asynccnt 0" ::: "memory");
        __syncthreads();

        // --- compute: 4 K-steps of 32, 2x2 fragments ---
        #pragma unroll
        for (int kc0 = 0; kc0 < KC; kc0 += 32) {
            V16U a0, a1, b0, b1;
            const unsigned short* a0p = arow0 + kc + kc0;
            const unsigned short* a1p = arow1 + kc + kc0;
            a0.q[0] = *(const u32x4*)(a0p);
            a0.q[1] = *(const u32x4*)(a0p + 16);
            a1.q[0] = *(const u32x4*)(a1p);
            a1.q[1] = *(const u32x4*)(a1p + 16);
            const unsigned short* b0p = &smemB[(wn * 32 + l16) * KC + kc0 + (hi ? 8 : 0)];
            const unsigned short* b1p = &smemB[(wn * 32 + 16 + l16) * KC + kc0 + (hi ? 8 : 0)];
            b0.q[0] = ((const u32x4*)b0p)[0];
            b0.q[1] = ((const u32x4*)b0p)[2];
            b1.q[0] = ((const u32x4*)b1p)[0];
            b1.q[1] = ((const u32x4*)b1p)[2];
            acc[0][0] = wmma_bf16(a0, b0, acc[0][0]);
            acc[0][1] = wmma_bf16(a0, b1, acc[0][1]);
            acc[1][0] = wmma_bf16(a1, b0, acc[1][0]);
            acc[1][1] = wmma_bf16(a1, b1, acc[1][1]);
        }
    }

    float w0 = dw[0], w1 = dw[1], bb = db[0];

    #pragma unroll
    for (int nf = 0; nf < 2; ++nf) {
        int j = n0w + nf * 16 + l16;
        f32x4 bj = *(const f32x4*)(boxes + 4 * (size_t)j);
        float aj = (bj.z - bj.x) * (bj.w - bj.y);
        float sqj = sq[j];
        #pragma unroll
        for (int mf = 0; mf < 2; ++mf) {
            #pragma unroll
            for (int r = 0; r < 8; ++r) {
                int i = m0w + mf * 16 + r + (hi ? 8 : 0);
                // cdist (zero-safe sqrt)
                float d2 = sq[i] + sqj - 2.0f * acc[mf][nf][r];
                float dv = (d2 > 0.0f) ? sqrtf(d2) : 0.0f;
                // GIoU
                f32x4 bi = *(const f32x4*)(boxes + 4 * (size_t)i);
                float ai = (bi.z - bi.x) * (bi.w - bi.y);
                float ltx = fmaxf(bi.x, bj.x), lty = fmaxf(bi.y, bj.y);
                float rbx = fminf(bi.z, bj.z), rby = fminf(bi.w, bj.w);
                float iw = fmaxf(rbx - ltx, 0.0f), ih = fmaxf(rby - lty, 0.0f);
                float inter = iw * ih;
                float uni = ai + aj - inter;
                float iou = inter / uni;
                float cx1 = fminf(bi.x, bj.x), cy1 = fminf(bi.y, bj.y);
                float cx2 = fmaxf(bi.z, bj.z), cy2 = fmaxf(bi.w, bj.w);
                float cw = fmaxf(cx2 - cx1, 0.0f), ch = fmaxf(cy2 - cy1, 0.0f);
                float areac = cw * ch;
                float giou = iou - (areac - uni) / areac;
                float dg = (giou + 1.0f) * 0.5f;
                float lg = w0 * dg + w1 * dv + bb;
                out[(size_t)i * NN + j] = 1.0f / (1.0f + expf(-lg));
            }
        }
    }
}

// ---------------------------------------------------------------------------
// 5) head layer 1: h[hd] = relu(emb @ w1[hd]^T + b1[hd]),  bf16 WMMA.
//    One 32(n) x 32(kk) tile per wave (2x2 fragments), K = 1024.
//    7 * 128 * 16 = 14336 waves -> 1792 blocks.
// ---------------------------------------------------------------------------
__global__ __launch_bounds__(256) void k_head_l1(const unsigned short* __restrict__ emb,   // [4096][1024]
                                                  const unsigned short* __restrict__ w1bf, // [7][512][1024]
                                                  const float* __restrict__ b1,            // [7][512]
                                                  unsigned short* __restrict__ h) {        // [7][4096][512] bf16
    int wave = threadIdx.x >> 5;
    int lane = threadIdx.x & 31;
    int l16  = lane & 15;
    int hi   = lane >> 4;

    int tile = blockIdx.x * 8 + wave;       // 14336 tiles
    int hd   = tile / 2048;                 // 128 * 16 tiles per head
    int rem  = tile % 2048;
    int n0   = (rem >> 4) * 32;
    int kk0  = (rem & 15) * 32;

    const unsigned short* arow0 = emb + (size_t)(n0 + l16) * EE + (hi ? 8 : 0);
    const unsigned short* arow1 = emb + (size_t)(n0 + 16 + l16) * EE + (hi ? 8 : 0);
    const unsigned short* brow0 = w1bf + ((size_t)hd * HOUT + kk0 + l16) * EE + (hi ? 8 : 0);
    const unsigned short* brow1 = w1bf + ((size_t)hd * HOUT + kk0 + 16 + l16) * EE + (hi ? 8 : 0);

    v8f acc[2][2] = {};
    #pragma unroll 2
    for (int e0 = 0; e0 < EE; e0 += 32) {
        V16U a0, a1, b0, b1;
        a0.q[0] = *(const u32x4*)(arow0 + e0);
        a0.q[1] = *(const u32x4*)(arow0 + e0 + 16);
        a1.q[0] = *(const u32x4*)(arow1 + e0);
        a1.q[1] = *(const u32x4*)(arow1 + e0 + 16);
        b0.q[0] = *(const u32x4*)(brow0 + e0);
        b0.q[1] = *(const u32x4*)(brow0 + e0 + 16);
        b1.q[0] = *(const u32x4*)(brow1 + e0);
        b1.q[1] = *(const u32x4*)(brow1 + e0 + 16);
        acc[0][0] = wmma_bf16(a0, b0, acc[0][0]);
        acc[0][1] = wmma_bf16(a0, b1, acc[0][1]);
        acc[1][0] = wmma_bf16(a1, b0, acc[1][0]);
        acc[1][1] = wmma_bf16(a1, b1, acc[1][1]);
    }

    #pragma unroll
    for (int nf = 0; nf < 2; ++nf) {
        int kk = kk0 + nf * 16 + l16;
        float bias = b1[(size_t)hd * HOUT + kk];
        #pragma unroll
        for (int mf = 0; mf < 2; ++mf) {
            #pragma unroll
            for (int r = 0; r < 8; ++r) {
                int n = n0 + mf * 16 + r + (hi ? 8 : 0);
                float v = acc[mf][nf][r] + bias;
                v = (v > 0.0f) ? v : 0.0f;
                h[((size_t)hd * NN + n) * HOUT + kk] = f2bf(v);
            }
        }
    }
}

// ---------------------------------------------------------------------------
// 6) head layer 2: tiny (output dims 2..7) -> plain VALU dot products
// ---------------------------------------------------------------------------
struct HeadArgs {
    const float* w2[7];
    const float* b2[7];
};

__global__ __launch_bounds__(256) void k_head_l2(const unsigned short* __restrict__ h,
                                                  HeadArgs ha,
                                                  float* __restrict__ out) {
    int idx = blockIdx.x * 256 + threadIdx.x;           // 4096 * 30
    if (idx >= NN * 30) return;
    int n = idx / 30;
    int r = idx % 30;
    const int od[7]    = {4, 4, 4, 2, 4, 7, 5};
    const int start[7] = {0, 4, 8, 12, 14, 18, 25};
    const size_t ooff[7] = {16777216ull, 16793600ull, 16809984ull, 16826368ull,
                            16834560ull, 16850944ull, 16879616ull};
    int head = 0;
    while (head < 6 && r >= start[head + 1]) head++;
    int o = r - start[head];

    const unsigned short* hrow = h + ((size_t)head * NN + n) * HOUT;
    const float* w = ha.w2[head] + (size_t)o * HOUT;
    float acc = 0.0f;
    #pragma unroll 8
    for (int k = 0; k < HOUT; ++k)
        acc += bf2f(hrow[k]) * w[k];
    acc += ha.b2[head][o];
    out[ooff[head] + (size_t)n * od[head] + o] = (acc > 0.0f) ? acc : 0.0f;
}

// ---------------------------------------------------------------------------
// launch
// ---------------------------------------------------------------------------
extern "C" void kernel_launch(void* const* d_in, const int* in_sizes, int n_in,
                              void* d_out, int out_size, void* d_ws, size_t ws_size,
                              hipStream_t stream) {
    const float* x      = (const float*)d_in[0];   // [4096,512]
    const float* boxes  = (const float*)d_in[1];   // [4096,4]
    const int*   seg    = (const int*)  d_in[2];   // [4096]
    const float* dw     = (const float*)d_in[4];   // [1,2]
    const float* db     = (const float*)d_in[5];   // [1]
    const float* w1     = (const float*)d_in[6];   // [7,512,1024]
    const float* b1     = (const float*)d_in[7];   // [7,512]

    HeadArgs ha;
    for (int hd = 0; hd < 7; ++hd) {
        ha.w2[hd] = (const float*)d_in[8 + 2 * hd];
        ha.b2[hd] = (const float*)d_in[9 + 2 * hd];
    }

    char* ws = (char*)d_ws;
    unsigned*       pool  = (unsigned*)(ws + 0);                         // 1,048,576 B
    float*          sq    = (float*)(ws + 1048576);                      //    16,384 B
    unsigned short* emb   = (unsigned short*)(ws + 1064960);             // 8,388,608 B
    unsigned short* w1bf  = (unsigned short*)(ws + 9453568);             // 7,340,032 B
    unsigned short* hbuf  = (unsigned short*)(ws + 16793600);            // 29,360,128 B

    float* out = (float*)d_out;

    k_zero_u32 <<<1024, 256, 0, stream>>>(pool, SS * DD);
    k_segmax   <<<NN,   256, 0, stream>>>(x, seg, pool);
    k_build_emb<<<NN,   256, 0, stream>>>(x, pool, seg, emb, sq);
    k_f2bf     <<<14336,256, 0, stream>>>(w1, w1bf, 7 * HOUT * EE);
    k_pair_sim <<<2048, 256, 0, stream>>>(emb, sq, boxes, dw, db, out);
    k_head_l1  <<<1792, 256, 0, stream>>>(emb, w1bf, b1, hbuf);
    k_head_l2  <<<480,  256, 0, stream>>>(hbuf, ha, out);
}
